// ToothFormer_59717225284064
// MI455X (gfx1250) — compile-verified, matrix operand order checked by reading, not run on hardware
//
#include <hip/hip_runtime.h>
#include <hip/hip_bf16.h>

typedef __attribute__((ext_vector_type(16))) _Float16 v16h;
typedef __attribute__((ext_vector_type(8)))  _Float16 v8h;
typedef __attribute__((ext_vector_type(8)))  float    v8f;

#define EMB   256
#define NH    8
#define HD    32
#define DEPTH 6
#define FF    512
#define MCTR  64
#define KNN   128
#define NC    26
#define BATCH 8
#define NPTS  32768

#define GF_RELU 1
#define GF_F16  2
#define GF_ADD  4

#if __has_builtin(__builtin_amdgcn_global_load_async_to_lds_b128) && \
    __has_builtin(__builtin_amdgcn_s_wait_asynccnt)
#define HAVE_ASYNC_LDS 1
typedef __attribute__((vector_size(16))) int vi4;
typedef vi4 __attribute__((address_space(1))) *gvi4p;   // pointer to global v4i
typedef vi4 __attribute__((address_space(3))) *lvi4p;   // pointer to LDS v4i
#else
#define HAVE_ASYNC_LDS 0
#endif

// ---------------------------------------------------------------------------
// WMMA fragment helpers (16x16x32 f16 -> f32).
// A-layout (16x32): lane r=lane&15 holds row; kb=(lane>=16)?8:0;
//   halves 0..7 -> K=kb..kb+7 (contiguous), halves 8..15 -> K=16+kb..
// B fragment for C=A*W^T loads W (N x Kd row-major) with the same mapping.
// Both are two 16-byte loads.
// ---------------------------------------------------------------------------
__device__ __forceinline__ v16h load_frag(const _Float16* p, int ld, int row0, int k0) {
  int lane = threadIdx.x & 31;
  int r    = lane & 15;
  int kb   = (lane >> 4) << 3;
  const _Float16* base = p + (size_t)(row0 + r) * ld + (k0 + kb);
  v8h lo = *(const v8h*)(base);
  v8h hi = *(const v8h*)(base + 16);
  v16h out;
#pragma unroll
  for (int i = 0; i < 8; ++i) { out[i] = lo[i]; out[i + 8] = hi[i]; }
  return out;
}

__device__ __forceinline__ v8f wmma16(v16h a, v16h b, v8f c) {
  return __builtin_amdgcn_wmma_f32_16x16x32_f16(false, a, false, b, (short)0, c, false, false);
}

// Stage a 64(N) x 32(K) f16 weight panel into LDS with 256 threads:
// thread t copies 16B: row = t>>2, segment = t&3.  Async when available.
__device__ __forceinline__ void stage_w(const _Float16* __restrict__ W, int ldw,
                                        int col0, int k0, _Float16* buf) {
  int t = threadIdx.x;
  int row = t >> 2;
  int seg = t & 3;
  const _Float16* src = W + (size_t)(col0 + row) * ldw + k0 + seg * 8;
  _Float16* dst = buf + row * 32 + seg * 8;
#if HAVE_ASYNC_LDS
  __builtin_amdgcn_global_load_async_to_lds_b128(
      (gvi4p)(void*)const_cast<_Float16*>(src),
      (lvi4p)(void*)dst, 0, 0);
#else
  *(v8h*)dst = *(const v8h*)src;
#endif
}

__device__ __forceinline__ void wait_stage() {
#if HAVE_ASYNC_LDS
  __builtin_amdgcn_s_wait_asynccnt(0);
#endif
  __syncthreads();
}

// ---------------------------------------------------------------------------
// Generic WMMA GEMM: C[M x Nd] = A[M x Kd] * W[Nd x Kd]^T (+bias)(+relu)
// Block = 256 threads = 8 waves; block tile 256(M) x 64(N); wave tile 32 x 64.
// W panel double-buffered in LDS via async copy.  Grid: (M/256, Nd/64).
// ---------------------------------------------------------------------------
__global__ void __launch_bounds__(256) k_gemm(
    const _Float16* __restrict__ A, int lda,
    const _Float16* __restrict__ W, int ldw,
    const float* __restrict__ bias,
    _Float16* __restrict__ C16, float* __restrict__ C32, int ldc,
    int Kd, int flags)
{
  __shared__ __align__(16) _Float16 wbuf[2][64][32];
  int wave = threadIdx.x >> 5;
  int row0 = blockIdx.x * 256 + wave * 32;
  int col0 = blockIdx.y * 64;
  v8f acc[2][4] = {};
  stage_w(W, ldw, col0, 0, &wbuf[0][0][0]);
  int nk = Kd >> 5;
  for (int i = 0; i < nk; ++i) {
    wait_stage();                                  // panel i resident
    if (i + 1 < nk) stage_w(W, ldw, col0, (i + 1) * 32, &wbuf[(i + 1) & 1][0][0]);
    const _Float16* bb = &wbuf[i & 1][0][0];
    v16h a0 = load_frag(A, lda, row0, i * 32);
    v16h a1 = load_frag(A, lda, row0 + 16, i * 32);
#pragma unroll
    for (int t = 0; t < 4; ++t) {
      v16h bf = load_frag(bb, 32, t * 16, 0);
      acc[0][t] = wmma16(a0, bf, acc[0][t]);
      acc[1][t] = wmma16(a1, bf, acc[1][t]);
    }
  }
  int lane = threadIdx.x & 31;
  int n    = lane & 15;
  int moff = (lane >> 4) << 3;
#pragma unroll
  for (int rt = 0; rt < 2; ++rt) {
#pragma unroll
    for (int t = 0; t < 4; ++t) {
      int col = col0 + t * 16 + n;
      float bv = bias ? bias[col] : 0.0f;
#pragma unroll
      for (int vr = 0; vr < 8; ++vr) {
        int row = row0 + rt * 16 + moff + vr;
        float v = acc[rt][t][vr] + bv;
        if (flags & GF_RELU) v = fmaxf(v, 0.0f);
        if (flags & GF_F16)      C16[(size_t)row * ldc + col] = (_Float16)v;
        else if (flags & GF_ADD) C32[(size_t)row * ldc + col] += v;
        else                     C32[(size_t)row * ldc + col] = v;
      }
    }
  }
}

// ---------------------------------------------------------------------------
// Patch-embed layer 3 fused with max-over-K: one block = one center (128 rows)
// x 64 cols.  x[center,col] = max_k(F2 @ W3^T) + b3 + pe.
// ---------------------------------------------------------------------------
__global__ void __launch_bounds__(256) k_gemm_maxpool(
    const _Float16* __restrict__ F2,
    const _Float16* __restrict__ W3,
    const float* __restrict__ b3,
    const float* __restrict__ pe,
    float* __restrict__ x)
{
  __shared__ __align__(16) _Float16 wbuf[2][64][32];
  __shared__ float colmax[8][64];
  int wave = threadIdx.x >> 5;
  int lane = threadIdx.x & 31;
  int center = blockIdx.x;
  int col0 = blockIdx.y * 64;
  int row0 = center * 128 + wave * 16;
  v8f acc[4] = {};
  stage_w(W3, 128, col0, 0, &wbuf[0][0][0]);
  for (int i = 0; i < 4; ++i) {
    wait_stage();
    if (i < 3) stage_w(W3, 128, col0, (i + 1) * 32, &wbuf[(i + 1) & 1][0][0]);
    const _Float16* bb = &wbuf[i & 1][0][0];
    v16h a = load_frag(F2, 128, row0, i * 32);
#pragma unroll
    for (int t = 0; t < 4; ++t) {
      v16h bf = load_frag(bb, 32, t * 16, 0);
      acc[t] = wmma16(a, bf, acc[t]);
    }
  }
  int n = lane & 15;
#pragma unroll
  for (int t = 0; t < 4; ++t) {
    float m8 = acc[t][0];
#pragma unroll
    for (int vr = 1; vr < 8; ++vr) m8 = fmaxf(m8, acc[t][vr]);
    float cr = __shfl_xor(m8, 16, 32);   // combine lane halves (m and m+8 groups)
    m8 = fmaxf(m8, cr);
    if (lane < 16) colmax[wave][t * 16 + n] = m8;
  }
  __syncthreads();
  if (threadIdx.x < 64) {
    int col = col0 + threadIdx.x;
    float m = colmax[0][threadIdx.x];
#pragma unroll
    for (int w = 1; w < 8; ++w) m = fmaxf(m, colmax[w][threadIdx.x]);
    x[(size_t)center * EMB + col] = m + b3[col] + pe[(center & 63) * EMB + col];
  }
}

// ---------------------------------------------------------------------------
// Exact kNN via 4-pass radix select on float bits of d^2, fused with
// MLP layer 1 (3->64, relu).  One block per center.
// ---------------------------------------------------------------------------
__global__ void k_knn_embed1(const float* __restrict__ xyz,
                             const float* __restrict__ w1,
                             const float* __restrict__ b1,
                             _Float16* __restrict__ F1)
{
  __shared__ unsigned hist[256];
  __shared__ unsigned s_sel, s_want, s_cnt;
  __shared__ float s_ctr[3];
  int c = blockIdx.x;
  int b = c >> 6, m = c & 63;
  int tid = threadIdx.x;
  if (tid == 0) {
    int ci = (m * (NPTS - 1)) / (MCTR - 1);  // linspace index
    const float* cp = xyz + ((size_t)b * NPTS + ci) * 3;
    s_ctr[0] = cp[0]; s_ctr[1] = cp[1]; s_ctr[2] = cp[2];
    s_cnt = 0u;
  }
  __syncthreads();
  float cx = s_ctr[0], cy = s_ctr[1], cz = s_ctr[2];
  const float* px = xyz + (size_t)b * NPTS * 3;

  unsigned prefix = 0, want = KNN;
  for (int pass = 0; pass < 4; ++pass) {
    int shift = 24 - 8 * pass;
    hist[tid] = 0u;
    __syncthreads();
    for (int i = tid; i < NPTS; i += 256) {
      float dx = px[i * 3] - cx, dy = px[i * 3 + 1] - cy, dz = px[i * 3 + 2] - cz;
      unsigned key = __float_as_uint(dx * dx + dy * dy + dz * dz);
      bool ok = (pass == 0) || ((key >> (shift + 8)) == prefix);
      if (ok) atomicAdd(&hist[(key >> shift) & 255], 1u);
    }
    __syncthreads();
    if (tid == 0) {
      unsigned cum = 0, sel = 255, w = want;
      for (int j = 0; j < 256; ++j) {
        unsigned h = hist[j];
        if (cum + h >= want) { sel = (unsigned)j; w = want - cum; break; }
        cum += h;
      }
      s_sel = sel; s_want = w;
    }
    __syncthreads();
    prefix = (prefix << 8) | s_sel;
    want = s_want;
    __syncthreads();
  }
  unsigned thresh = prefix;   // K-th smallest key

  for (int i = tid; i < NPTS; i += 256) {
    float dx = px[i * 3] - cx, dy = px[i * 3 + 1] - cy, dz = px[i * 3 + 2] - cz;
    unsigned key = __float_as_uint(dx * dx + dy * dy + dz * dz);
    if (key <= thresh) {
      unsigned slot = atomicAdd(&s_cnt, 1u);
      if (slot < KNN) {
        _Float16* dst = F1 + ((size_t)c * KNN + slot) * 64;
#pragma unroll 4
        for (int j = 0; j < 64; ++j) {
          float v = w1[j * 3] * dx + w1[j * 3 + 1] * dy + w1[j * 3 + 2] * dz + b1[j];
          dst[j] = (_Float16)fmaxf(v, 0.0f);
        }
      }
    }
  }
}

// ---------------------------------------------------------------------------
// LayerNorm over EMB=256; one block per token, writes f16.
// ---------------------------------------------------------------------------
__global__ void k_ln(const float* __restrict__ x, const float* __restrict__ g,
                     const float* __restrict__ bta, _Float16* __restrict__ h)
{
  __shared__ float red[256];
  int t = blockIdx.x;
  float v = x[(size_t)t * EMB + threadIdx.x];
  red[threadIdx.x] = v; __syncthreads();
  for (int s = 128; s > 0; s >>= 1) {
    if (threadIdx.x < s) red[threadIdx.x] += red[threadIdx.x + s];
    __syncthreads();
  }
  float mean = red[0] * (1.0f / EMB);
  __syncthreads();
  float d = v - mean;
  red[threadIdx.x] = d * d; __syncthreads();
  for (int s = 128; s > 0; s >>= 1) {
    if (threadIdx.x < s) red[threadIdx.x] += red[threadIdx.x + s];
    __syncthreads();
  }
  float r = rsqrtf(red[0] * (1.0f / EMB) + 1e-5f);
  h[(size_t)t * EMB + threadIdx.x] = (_Float16)(d * r * g[threadIdx.x] + bta[threadIdx.x]);
}

// ---------------------------------------------------------------------------
// Attention for one (batch, head): scores = QK^T*scale, softmax, O = att*V.
// Block = 128 threads (4 waves); all intermediates in LDS.
// ---------------------------------------------------------------------------
__global__ void k_attn(const _Float16* __restrict__ qkv, _Float16* __restrict__ obuf)
{
  __shared__ float    sc[64][64];
  __shared__ __align__(16) _Float16 att[64][64];
  __shared__ __align__(16) _Float16 vt[32][64];
  int bh = blockIdx.x;
  int b = bh >> 3, h = bh & 7;
  int wave = threadIdx.x >> 5;
  int lane = threadIdx.x & 31;
  int m0 = wave * 16;
  int rowbase = b * 64;

  v8f acc[4] = {};
  {
    v16h a = load_frag(qkv, 3 * EMB, rowbase + m0, h * HD);
#pragma unroll
    for (int t = 0; t < 4; ++t) {
      v16h kf = load_frag(qkv, 3 * EMB, rowbase + t * 16, EMB + h * HD);
      acc[t] = wmma16(a, kf, acc[t]);
    }
  }
  int n = lane & 15;
  int moff = (lane >> 4) << 3;
  const float scale = 0.17677669529663687f;  // 1/sqrt(32)
#pragma unroll
  for (int t = 0; t < 4; ++t)
#pragma unroll
    for (int vr = 0; vr < 8; ++vr)
      sc[m0 + moff + vr][t * 16 + n] = acc[t][vr] * scale;

  for (int i = threadIdx.x; i < HD * 64; i += blockDim.x) {   // vt[n][k] = V[k][n]
    int d = i >> 6, tt = i & 63;
    vt[d][tt] = qkv[(size_t)(rowbase + tt) * (3 * EMB) + 2 * EMB + h * HD + d];
  }
  __syncthreads();
  if (threadIdx.x < 64) {
    int r = threadIdx.x;
    float mx = sc[r][0];
    for (int j = 1; j < 64; ++j) mx = fmaxf(mx, sc[r][j]);
    float s = 0.0f;
    for (int j = 0; j < 64; ++j) { float e = __expf(sc[r][j] - mx); sc[r][j] = e; s += e; }
    float inv = 1.0f / s;
    for (int j = 0; j < 64; ++j) att[r][j] = (_Float16)(sc[r][j] * inv);
  }
  __syncthreads();

  v8f acc2[2] = {};
  for (int k0 = 0; k0 < 64; k0 += 32) {
    v16h a2 = load_frag(&att[0][0], 64, m0, k0);
#pragma unroll
    for (int t = 0; t < 2; ++t) {
      v16h b2 = load_frag(&vt[0][0], 64, t * 16, k0);
      acc2[t] = wmma16(a2, b2, acc2[t]);
    }
  }
#pragma unroll
  for (int t = 0; t < 2; ++t)
#pragma unroll
    for (int vr = 0; vr < 8; ++vr)
      obuf[(size_t)(rowbase + m0 + moff + vr) * EMB + h * HD + t * 16 + n] = (_Float16)acc2[t][vr];
}

// ---------------------------------------------------------------------------
// Utility kernels
// ---------------------------------------------------------------------------
__global__ void k_cvt(const float* __restrict__ src, _Float16* __restrict__ dst, int nelem) {
  int i = blockIdx.x * 256 + threadIdx.x;
  if (i < nelem) dst[i] = (_Float16)src[i];
}

__global__ void k_pad_h2(const float* __restrict__ w, const float* __restrict__ bs,
                         _Float16* __restrict__ wp, float* __restrict__ bp) {
  int i = blockIdx.x * 256 + threadIdx.x;
  if (i < 64 * EMB) wp[i] = (i < NC * EMB) ? (_Float16)w[i] : (_Float16)0.0f;
  if (i < 64) bp[i] = (i < NC) ? bs[i] : 0.0f;
}

// Per-point nearest center + gather of per-token head output (26 classes).
__global__ void k_assign(const float* __restrict__ xyz, const float* __restrict__ t3,
                         float* __restrict__ out)
{
  __shared__ float ctr[64][3];
  const int blocksPerB = NPTS / 256;
  int b  = blockIdx.x / blocksPerB;
  int n0 = (blockIdx.x % blocksPerB) * 256;
  if (threadIdx.x < 64) {
    int m = threadIdx.x;
    int ci = (m * (NPTS - 1)) / (MCTR - 1);
    const float* cp = xyz + ((size_t)b * NPTS + ci) * 3;
    ctr[m][0] = cp[0]; ctr[m][1] = cp[1]; ctr[m][2] = cp[2];
  }
  __syncthreads();
  int n = n0 + threadIdx.x;
  const float* pp = xyz + ((size_t)b * NPTS + n) * 3;
  float x = pp[0], y = pp[1], z = pp[2];
  int best = 0; float bd = 3.4e38f;
#pragma unroll 8
  for (int m = 0; m < 64; ++m) {
    float dx = x - ctr[m][0], dy = y - ctr[m][1], dz = z - ctr[m][2];
    float d = dx * dx + dy * dy + dz * dz;
    if (d < bd) { bd = d; best = m; }
  }
  const float* src = t3 + (size_t)(b * 64 + best) * 64;
  float* dst = out + ((size_t)b * NPTS + n) * NC;
#pragma unroll
  for (int j = 0; j < NC; ++j) dst[j] = src[j];
}

// ---------------------------------------------------------------------------
extern "C" void kernel_launch(void* const* d_in, const int* in_sizes, int n_in,
                              void* d_out, int out_size, void* d_ws, size_t ws_size,
                              hipStream_t stream) {
  const float* xyz   = (const float*)d_in[0];
  const float* pe    = (const float*)d_in[1];
  const float* pm_w1 = (const float*)d_in[2];  const float* pm_b1 = (const float*)d_in[3];
  const float* pm_w2 = (const float*)d_in[4];  const float* pm_b2 = (const float*)d_in[5];
  const float* pm_w3 = (const float*)d_in[6];  const float* pm_b3 = (const float*)d_in[7];
  const float* ln1g  = (const float*)d_in[8];  const float* ln1b  = (const float*)d_in[9];
  const float* wqkv  = (const float*)d_in[10]; const float* bqkv  = (const float*)d_in[11];
  const float* wo    = (const float*)d_in[12]; const float* bo    = (const float*)d_in[13];
  const float* ln2g  = (const float*)d_in[14]; const float* ln2b  = (const float*)d_in[15];
  const float* fw1   = (const float*)d_in[16]; const float* fb1   = (const float*)d_in[17];
  const float* fw2   = (const float*)d_in[18]; const float* fb2   = (const float*)d_in[19];
  const float* projw = (const float*)d_in[20]; const float* projb = (const float*)d_in[21];
  const float* h1w   = (const float*)d_in[22]; const float* h1b   = (const float*)d_in[23];
  const float* h2w   = (const float*)d_in[24]; const float* h2b   = (const float*)d_in[25];
  float* out = (float*)d_out;

  char* ws = (char*)d_ws;
  size_t off = 0;
  auto carve = [&](size_t bytes) -> char* {
    char* p = ws + off; off = (off + bytes + 255) & ~(size_t)255; return p;
  };
  const int NTOK = BATCH * MCTR;            // 512
  const int NPAT = NTOK * KNN;              // 65536 patch points

  _Float16* w2h   = (_Float16*)carve(128 * 64 * 2);
  _Float16* w3h   = (_Float16*)carve(256 * 128 * 2);
  _Float16* wqkvh = (_Float16*)carve((size_t)DEPTH * 768 * 256 * 2);
  _Float16* woh   = (_Float16*)carve((size_t)DEPTH * 256 * 256 * 2);
  _Float16* f1wh  = (_Float16*)carve((size_t)DEPTH * 512 * 256 * 2);
  _Float16* f2wh  = (_Float16*)carve((size_t)DEPTH * 256 * 512 * 2);
  _Float16* projh = (_Float16*)carve(256 * 256 * 2);
  _Float16* h1h   = (_Float16*)carve(256 * 256 * 2);
  _Float16* h2ph  = (_Float16*)carve(64 * 256 * 2);
  float*    h2bp  = (float*)   carve(64 * 4);
  _Float16* F1    = (_Float16*)carve((size_t)NPAT * 64 * 2);
  _Float16* F2    = (_Float16*)carve((size_t)NPAT * 128 * 2);
  float*    xbuf  = (float*)   carve((size_t)NTOK * EMB * 4);
  _Float16* xh    = (_Float16*)carve((size_t)NTOK * EMB * 2);
  _Float16* hbuf  = (_Float16*)carve((size_t)NTOK * EMB * 2);
  _Float16* qkvb  = (_Float16*)carve((size_t)NTOK * 768 * 2);
  _Float16* obuf  = (_Float16*)carve((size_t)NTOK * EMB * 2);
  _Float16* ffnb  = (_Float16*)carve((size_t)NTOK * FF * 2);
  _Float16* t1    = (_Float16*)carve((size_t)NTOK * EMB * 2);
  _Float16* t2    = (_Float16*)carve((size_t)NTOK * EMB * 2);
  float*    t3    = (float*)   carve((size_t)NTOK * 64 * 4);
  (void)ws_size; (void)n_in; (void)in_sizes; (void)out_size;

  auto cvt = [&](const float* s, _Float16* d, int n) {
    k_cvt<<<(n + 255) / 256, 256, 0, stream>>>(s, d, n);
  };
  cvt(pm_w2, w2h, 128 * 64);
  cvt(pm_w3, w3h, 256 * 128);
  cvt(wqkv,  wqkvh, DEPTH * 768 * 256);
  cvt(wo,    woh,   DEPTH * 256 * 256);
  cvt(fw1,   f1wh,  DEPTH * 512 * 256);
  cvt(fw2,   f2wh,  DEPTH * 256 * 512);
  cvt(projw, projh, 256 * 256);
  cvt(h1w,   h1h,   256 * 256);
  k_pad_h2<<<64, 256, 0, stream>>>(h2w, h2b, h2ph, h2bp);

  // kNN + layer1
  k_knn_embed1<<<NTOK, 256, 0, stream>>>(xyz, pm_w1, pm_b1, F1);
  // layer2: F2 = relu(F1 @ W2^T + b2)   (65536x128, Kd=64)
  k_gemm<<<dim3(NPAT / 256, 2), 256, 0, stream>>>(F1, 64, w2h, 64, pm_b2,
                                                  F2, nullptr, 128, 64, GF_RELU | GF_F16);
  // layer3 + maxpool + pe -> tokens x (512x256 f32)
  k_gemm_maxpool<<<dim3(NTOK, 4), 256, 0, stream>>>(F2, w3h, pm_b3, pe, xbuf);

  for (int l = 0; l < DEPTH; ++l) {
    k_ln<<<NTOK, 256, 0, stream>>>(xbuf, ln1g + l * EMB, ln1b + l * EMB, hbuf);
    k_gemm<<<dim3(NTOK / 256, 12), 256, 0, stream>>>(hbuf, 256,
        wqkvh + (size_t)l * 768 * 256, 256, bqkv + l * 768,
        qkvb, nullptr, 768, 256, GF_F16);
    k_attn<<<BATCH * NH, 128, 0, stream>>>(qkvb, obuf);
    k_gemm<<<dim3(NTOK / 256, 4), 256, 0, stream>>>(obuf, 256,
        woh + (size_t)l * 256 * 256, 256, bo + l * EMB,
        nullptr, xbuf, 256, 256, GF_ADD);
    k_ln<<<NTOK, 256, 0, stream>>>(xbuf, ln2g + l * EMB, ln2b + l * EMB, hbuf);
    k_gemm<<<dim3(NTOK / 256, 8), 256, 0, stream>>>(hbuf, 256,
        f1wh + (size_t)l * 512 * 256, 256, fb1 + l * FF,
        ffnb, nullptr, 512, 256, GF_RELU | GF_F16);
    k_gemm<<<dim3(NTOK / 256, 4), 256, 0, stream>>>(ffnb, 512,
        f2wh + (size_t)l * 256 * 512, 512, fb2 + l * EMB,
        nullptr, xbuf, 256, 512, GF_ADD);
  }

  // per-token head chain (points sharing a center share the result)
  k_cvt<<<(NTOK * EMB + 255) / 256, 256, 0, stream>>>(xbuf, xh, NTOK * EMB);
  k_gemm<<<dim3(NTOK / 256, 4), 256, 0, stream>>>(xh, 256, projh, 256, projb,
                                                  t1, nullptr, 256, 256, GF_F16);
  k_gemm<<<dim3(NTOK / 256, 4), 256, 0, stream>>>(t1, 256, h1h, 256, h1b,
                                                  t2, nullptr, 256, 256, GF_RELU | GF_F16);
  k_gemm<<<dim3(NTOK / 256, 1), 256, 0, stream>>>(t2, 256, h2ph, 256, h2bp,
                                                  nullptr, t3, 64, 256, 0);
  // nearest-center gather -> output (B,N,26)
  k_assign<<<BATCH * NPTS / 256, 256, 0, stream>>>(xyz, t3, out);
}
